// SpectralLowRankOperator_37048387895879
// MI455X (gfx1250) — matrix operand order repaired and checked by reading, not compiled, not run on hardware
//
#include <hip/hip_runtime.h>

// Problem constants (from reference)
#define BDIM 4
#define SDIM 8192
#define DDIM 1024
#define ODIM 1024
#define KDIM 256
#define RDIM 64

typedef __attribute__((ext_vector_type(16))) __bf16        bf16x16;
typedef __attribute__((ext_vector_type(8)))  float         f32x8;
typedef __attribute__((ext_vector_type(4)))  float         f32x4;
typedef __attribute__((ext_vector_type(4)))  unsigned int  u32x4;

union ABf {
  bf16x16 bf;
  u32x4   u4[2];
};

static __device__ __forceinline__ unsigned short f2bf_bits(float f) {
  union { __bf16 h; unsigned short u; } c;
  c.h = (__bf16)f;
  return c.u;
}

// ---------------------------------------------------------------------------
// Step 1: gram = raw^T raw + 1e-4 I   (raw = mode_to_rank [K=256, R=64])
// ---------------------------------------------------------------------------
__global__ void k_gram(const float* __restrict__ raw, float* __restrict__ gram) {
  int e = blockIdx.x * blockDim.x + threadIdx.x;   // 4096 entries
  int r = e >> 6, c = e & 63;
  float s = 0.f;
  for (int k = 0; k < KDIM; ++k)
    s += raw[k * RDIM + r] * raw[k * RDIM + c];
  if (r == c) s += 1e-4f;
  gram[e] = s;
}

// ---------------------------------------------------------------------------
// Step 2: Cholesky of gram (in LDS) + forward-substitution inverse -> invL
// ---------------------------------------------------------------------------
__global__ void k_cholinv(const float* __restrict__ gram, float* __restrict__ invL) {
  __shared__ float g[RDIM][RDIM + 1];
  const int i = threadIdx.x;           // 0..63
  for (int c = 0; c < RDIM; ++c) g[i][c] = gram[i * RDIM + c];
  __syncthreads();

  for (int j = 0; j < RDIM; ++j) {
    float gjj = g[j][j];
    __syncthreads();
    float d = sqrtf(gjj);
    if (i == j)      g[i][j] = d;
    else if (i > j)  g[i][j] = g[i][j] / d;
    __syncthreads();
    if (i > j) {
      float lij = g[i][j];
      for (int c = j + 1; c <= i; ++c) g[i][c] -= lij * g[c][j];
    }
    __syncthreads();
  }

  const int c = i;
  float xcol[RDIM];
  for (int r = 0; r < RDIM; ++r) xcol[r] = 0.f;
  xcol[c] = 1.0f / g[c][c];
  for (int r = c + 1; r < RDIM; ++r) {
    float s = 0.f;
    for (int j = c; j < r; ++j) s += g[r][j] * xcol[j];
    xcol[r] = -s / g[r][r];
  }
  for (int r = 0; r < RDIM; ++r) invL[r * RDIM + c] = xcol[r];
}

// ---------------------------------------------------------------------------
// Step 3: q = raw @ invL^T, signs = sign(sum_k q*raw), mtr = q * signs
// ---------------------------------------------------------------------------
__global__ void k_mtr(const float* __restrict__ raw, const float* __restrict__ invL,
                      float* __restrict__ mtr) {
  __shared__ float sL[RDIM * RDIM];
  __shared__ float sig[RDIM];
  const int tid = threadIdx.x;        // 0..255 == k
  for (int t = tid; t < RDIM * RDIM; t += 256) sL[t] = invL[t];
  if (tid < RDIM) sig[tid] = 0.f;
  __syncthreads();

  float rr[RDIM], qr[RDIM];
  for (int j = 0; j < RDIM; ++j) rr[j] = raw[tid * RDIM + j];
  for (int r = 0; r < RDIM; ++r) {
    float s = 0.f;
    for (int j = 0; j <= r; ++j) s += rr[j] * sL[r * RDIM + j];
    qr[r] = s;
    atomicAdd(&sig[r], s * rr[r]);
  }
  __syncthreads();
  for (int r = 0; r < RDIM; ++r) {
    float sg = (sig[r] >= 0.f) ? 1.f : -1.f;
    mtr[tid * RDIM + r] = qr[r] * sg;
  }
}

// ---------------------------------------------------------------------------
// Step 4: W1t[b][r][d] = rc[b][r] * sum_k inB[k][d]*sc[b][k]*mtr[k][r]  (bf16)
// ---------------------------------------------------------------------------
__global__ void k_w1(const float* __restrict__ inB, const float* __restrict__ sc,
                     const float* __restrict__ rc, const float* __restrict__ mtr,
                     unsigned short* __restrict__ w1t) {
  const int d = blockIdx.x;
  const int t = threadIdx.x;
  const int b = t >> 6, r = t & 63;
  float acc = 0.f;
  for (int k = 0; k < KDIM; ++k)
    acc += inB[k * DDIM + d] * sc[b * KDIM + k] * mtr[k * RDIM + r];
  acc *= rc[b * RDIM + r];
  w1t[((size_t)(b * RDIM + r)) * DDIM + d] = f2bf_bits(acc);
}

// Step 5: W2t[o][r] = bf16(output_basis[r][o])
__global__ void k_w2(const float* __restrict__ outB, unsigned short* __restrict__ w2t) {
  int idx = blockIdx.x * 256 + threadIdx.x;
  int o = idx >> 6, r = idx & 63;
  w2t[o * RDIM + r] = f2bf_bits(outB[r * ODIM + o]);
}

// ---------------------------------------------------------------------------
// Main fused kernel, v3: 8 waves / 256 threads per block, 128 rows per block.
//  - W1 (batch slice) and W2 staged once per block into padded LDS.
//  - x streamed HBM->LDS with double-buffered global_load_async_to_lds_b128
//    (ASYNCcnt-tracked; s_wait_asynccnt + barrier per chunk).
//  - WMMAs consume LDS operands only; out stored non-temporally.
// LDS map (single block at offset 0):
//   [0)        W1: 64 rows x 1032 bf16 (pad 8)   = 132096 B
//   [132096)   W2: 1024 rows x 72 bf16 (pad 8)   = 147456 B
//   [279552)   X : 2 buffers x 128 rows x 36 f32 =  36864 B   (hid aliases buf0)
// ---------------------------------------------------------------------------
#define SW1_OFF   0
#define SW1_RB    2064                 /* 1032 bf16 row stride, bytes */
#define SW2_OFF   132096
#define SW2_RB    144                  /* 72 bf16 row stride, bytes  */
#define SX_OFF    279552
#define SX_RB     144                  /* 36 f32 row stride, bytes   */
#define SX_BUF    18432                /* 128 * 144                  */
#define SMEM_SZ   316416

__global__ __launch_bounds__(256) void k_main(const float* __restrict__ x,
                                              const unsigned short* __restrict__ w1t,
                                              const unsigned short* __restrict__ w2t,
                                              float* __restrict__ out) {
  __shared__ __align__(16) char smem[SMEM_SZ];
  const int tid   = threadIdx.x;
  const int lane  = tid & 31;
  const int wave  = tid >> 5;                  // 0..7
  const int l15   = lane & 15;
  const int hi    = lane >> 4;                 // 0 = lanes 0-15, 1 = lanes 16-31
  const int batch = blockIdx.y;
  const int rowBase = blockIdx.x * 128;        // block's first row in this batch

  // ---- stage W1 (this batch) and W2 into LDS ----
  const unsigned short* w1 = w1t + (size_t)batch * RDIM * DDIM;
  for (int i = tid; i < 64 * 128; i += 256) {          // 64 rows x 128 b128
    int r = i >> 7, s = i & 127;
    u32x4 v = *(const u32x4*)(w1 + r * 1024 + s * 8);
    *(u32x4*)(smem + SW1_OFF + r * SW1_RB + s * 16) = v;
  }
  for (int i = tid; i < 1024 * 8; i += 256) {          // 1024 rows x 8 b128
    int r = i >> 3, s = i & 7;
    u32x4 v = *(const u32x4*)(w2t + r * 64 + s * 8);
    *(u32x4*)(smem + SW2_OFF + r * SW2_RB + s * 16) = v;
  }

  // ---- async x chunk copier: chunk kt = cols [kt*32, kt*32+32) of 128 rows ----
  const float* xblk = x + ((size_t)batch * SDIM + rowBase) * DDIM;
  const int cprow  = tid >> 1;                 // 0..127: row this thread copies
  const int cphalf = tid & 1;                  // which 64B half of the 128B row chunk
  #define ASYNC_CHUNK(KT, BUF)                                                   \
    {                                                                            \
      const float*   gp   = xblk + (size_t)cprow * DDIM + (KT) * 32 + cphalf*16; \
      unsigned       lofs = SX_OFF + (BUF) * SX_BUF + cprow * SX_RB + cphalf*64; \
      asm volatile("global_load_async_to_lds_b128 %0, %1, off"                   \
                   :: "v"(lofs),      "v"(gp)      : "memory");                  \
      asm volatile("global_load_async_to_lds_b128 %0, %1, off"                   \
                   :: "v"(lofs + 16), "v"(gp + 4)  : "memory");                  \
      asm volatile("global_load_async_to_lds_b128 %0, %1, off"                   \
                   :: "v"(lofs + 32), "v"(gp + 8)  : "memory");                  \
      asm volatile("global_load_async_to_lds_b128 %0, %1, off"                   \
                   :: "v"(lofs + 48), "v"(gp + 12) : "memory");                  \
    }

  ASYNC_CHUNK(0, 0)
  asm volatile("s_wait_asynccnt 0x0" ::: "memory");
  __syncthreads();                             // W1/W2 staged + chunk 0 landed

  f32x8 acc1[4];
#pragma unroll
  for (int t = 0; t < 4; ++t)
#pragma unroll
    for (int e = 0; e < 8; ++e) acc1[t][e] = 0.f;

  const char* xlds_row = smem + SX_OFF + (size_t)(wave * 16 + l15) * SX_RB + hi * 32;

  for (int kt = 0; kt < DDIM / 32; ++kt) {
    const int buf = kt & 1;
    if (kt + 1 < DDIM / 32) ASYNC_CHUNK(kt + 1, buf ^ 1)

    // A fragment: 16 f32 from LDS chunk -> bf16
    const float* p = (const float*)(xlds_row + buf * SX_BUF);
    f32x4 c0 = *(const f32x4*)(p);
    f32x4 c1 = *(const f32x4*)(p + 4);
    f32x4 c2 = *(const f32x4*)(p + 16);
    f32x4 c3 = *(const f32x4*)(p + 20);
    ABf a;
    a.bf[0]  = (__bf16)c0[0]; a.bf[1]  = (__bf16)c0[1]; a.bf[2]  = (__bf16)c0[2]; a.bf[3]  = (__bf16)c0[3];
    a.bf[4]  = (__bf16)c1[0]; a.bf[5]  = (__bf16)c1[1]; a.bf[6]  = (__bf16)c1[2]; a.bf[7]  = (__bf16)c1[3];
    a.bf[8]  = (__bf16)c2[0]; a.bf[9]  = (__bf16)c2[1]; a.bf[10] = (__bf16)c2[2]; a.bf[11] = (__bf16)c2[3];
    a.bf[12] = (__bf16)c3[0]; a.bf[13] = (__bf16)c3[1]; a.bf[14] = (__bf16)c3[2]; a.bf[15] = (__bf16)c3[3];

    // B tiles from LDS-resident W1 (conflict-free padded rows)
    const int kbbB = (kt * 32 + hi * 8) * 2;   // byte offset in a W1 row
#pragma unroll
    for (int t = 0; t < 4; ++t) {
      const char* q = smem + SW1_OFF + (size_t)(t * 16 + l15) * SW1_RB + kbbB;
      ABf b;
      b.u4[0] = *(const u32x4*)(q);
      b.u4[1] = *(const u32x4*)(q + 32);
      acc1[t] = __builtin_amdgcn_wmma_f32_16x16x32_bf16(false, a.bf, false, b.bf,
                                                        (short)0, acc1[t], false, false);
    }

    asm volatile("s_wait_asynccnt 0x0" ::: "memory");  // next chunk landed
    __syncthreads();                                   // all waves done with 'buf'
  }

  // ---- hidden[16,64] -> LDS (aliases x buffer 0; barrier above protects) ----
  unsigned short (*hid)[16][72] = (unsigned short (*)[16][72])(smem + SX_OFF);
#pragma unroll
  for (int t = 0; t < 4; ++t)
#pragma unroll
    for (int v = 0; v < 8; ++v)
      hid[wave][v + hi * 8][t * 16 + l15] = f2bf_bits(acc1[t][v]);
  __syncthreads();

  // Re-gather hidden as two 16x32 A tiles (K = r dimension)
  ABf a0, a1;
  {
    const unsigned short* hp = &hid[wave][l15][0];
    const int kb = hi * 8;
    a0.u4[0] = *(const u32x4*)(hp + kb);
    a0.u4[1] = *(const u32x4*)(hp + kb + 16);
    a1.u4[0] = *(const u32x4*)(hp + kb + 32);
    a1.u4[1] = *(const u32x4*)(hp + kb + 48);
  }

  // ---- stage 2: out[16,1024] = hidden @ W2 (B tiles from LDS) ----
  float* orow = out + ((size_t)batch * SDIM + rowBase + wave * 16) * ODIM;
  for (int j = 0; j < ODIM / 16; ++j) {
    const char* q = smem + SW2_OFF + (size_t)(j * 16 + l15) * SW2_RB + hi * 16;
    ABf b0, b1;
    b0.u4[0] = *(const u32x4*)(q);
    b0.u4[1] = *(const u32x4*)(q + 32);
    b1.u4[0] = *(const u32x4*)(q + 64);
    b1.u4[1] = *(const u32x4*)(q + 96);
    f32x8 acc;
#pragma unroll
    for (int e = 0; e < 8; ++e) acc[e] = 0.f;
    acc = __builtin_amdgcn_wmma_f32_16x16x32_bf16(false, a0.bf, false, b0.bf, (short)0, acc, false, false);
    acc = __builtin_amdgcn_wmma_f32_16x16x32_bf16(false, a1.bf, false, b1.bf, (short)0, acc, false, false);
#pragma unroll
    for (int v = 0; v < 8; ++v)
      __builtin_nontemporal_store(acc[v], &orow[(size_t)(v + hi * 8) * ODIM + j * 16 + l15]);
  }
}

// ---------------------------------------------------------------------------
extern "C" void kernel_launch(void* const* d_in, const int* in_sizes, int n_in,
                              void* d_out, int out_size, void* d_ws, size_t ws_size,
                              hipStream_t stream) {
  (void)in_sizes; (void)n_in; (void)out_size; (void)ws_size;
  const float* x    = (const float*)d_in[0];   // [B,S,D]
  const float* sc   = (const float*)d_in[1];   // [B,K]
  const float* rc   = (const float*)d_in[2];   // [B,R]
  const float* inB  = (const float*)d_in[3];   // [K,D]
  const float* raw  = (const float*)d_in[4];   // [K,R]
  const float* outB = (const float*)d_in[5];   // [R,O]
  float* out = (float*)d_out;

  char* ws = (char*)d_ws;
  float*          gram = (float*)(ws + 0);                        // 16 KB
  float*          invL = (float*)(ws + 16384);                    // 16 KB
  float*          mtr  = (float*)(ws + 32768);                    // 64 KB
  unsigned short* w1t  = (unsigned short*)(ws + 98304);           // 512 KB [B][R][D] bf16
  unsigned short* w2t  = (unsigned short*)(ws + 98304 + 524288);  // 128 KB [O][R] bf16

  k_gram   <<<16, 256, 0, stream>>>(raw, gram);
  k_cholinv<<<1, 64, 0, stream>>>(gram, invL);
  k_mtr    <<<1, 256, 0, stream>>>(raw, invL, mtr);
  k_w1     <<<DDIM, 256, 0, stream>>>(inB, sc, rc, mtr, w1t);
  k_w2     <<<256, 256, 0, stream>>>(outB, w2t);
  dim3 grid(SDIM / 128, BDIM);
  k_main   <<<grid, 256, 0, stream>>>(x, w1t, w2t, out);
}
